// Grok1MoE_23261542875712
// MI455X (gfx1250) — compile-verified, matrix operand order checked by reading, not run on hardware
//
#include <hip/hip_runtime.h>
#include <hip/hip_bf16.h>

typedef __attribute__((ext_vector_type(16))) __bf16 v16bf;
typedef __attribute__((ext_vector_type(8)))  float  v8f;

#define T_TOK   2048
#define D_DIM   2048
#define E_EXP   8
#define I_DIM   2048
#define SOFTCAPF 30.0f

#define TM   128   // token tile per workgroup (8 waves x 16 rows)
#define TN   64    // output-column tile per workgroup
#define TK   32    // K chunk (matches wmma 16x16x32 bf16)
#define LDSK 40    // padded LDS K stride (bf16 elems): even, 20-bank stride

union FragBF { v16bf v; unsigned int u[8]; };

// Load a 16x32 bf16 operand fragment from an LDS tile with row stride LDSK.
// A-operand layout per ISA 7.12.2: lanes 0-15 row M=lane, lanes 16-31 same M,
// K halves split: VGPR j(0..3): K = hi*8 + 2j,2j+1 ; VGPR j(4..7): K += 16.
// The same mapping is used for the B operand against an [N][K] tile.
__device__ inline v16bf load_frag_b16(const __bf16* base, int lane) {
  const unsigned int* b32 = (const unsigned int*)base;
  const int row = lane & 15;
  const int hi  = (lane >> 4) & 1;
  FragBF f;
#pragma unroll
  for (int j = 0; j < 4; ++j) {
    f.u[j]     = b32[(row * LDSK + hi * 8 + 2 * j) >> 1];
    f.u[j + 4] = b32[(row * LDSK + 16 + hi * 8 + 2 * j) >> 1];
  }
  return f.v;
}

// ---------------- Router: logits -> softcap -> softmax -> top-2 combine ----
__global__ __launch_bounds__(256) void router_kernel(
    const float* __restrict__ x, const float* __restrict__ wg,
    float* __restrict__ combine) {
  const int lane = threadIdx.x & 31;
  const int wave = threadIdx.x >> 5;
  const int t = blockIdx.x * 8 + wave;
  if (t >= T_TOK) return;

  float acc[E_EXP];
#pragma unroll
  for (int e = 0; e < E_EXP; ++e) acc[e] = 0.f;
  for (int d = lane; d < D_DIM; d += 32) {
    const float xv = x[(size_t)t * D_DIM + d];
#pragma unroll
    for (int e = 0; e < E_EXP; ++e) acc[e] += xv * wg[(size_t)e * D_DIM + d];
  }
#pragma unroll
  for (int e = 0; e < E_EXP; ++e) {
#pragma unroll
    for (int off = 16; off > 0; off >>= 1)
      acc[e] += __shfl_xor(acc[e], off, 32);
  }
  // softcap + softmax (replicated per lane; lane 0 writes)
  float p[E_EXP];
  float mx = -1e30f;
#pragma unroll
  for (int e = 0; e < E_EXP; ++e) {
    p[e] = SOFTCAPF * tanhf(acc[e] * (1.0f / SOFTCAPF));
    mx = fmaxf(mx, p[e]);
  }
  float sum = 0.f;
#pragma unroll
  for (int e = 0; e < E_EXP; ++e) { p[e] = __expf(p[e] - mx); sum += p[e]; }
  const float inv = 1.0f / sum;
#pragma unroll
  for (int e = 0; e < E_EXP; ++e) p[e] *= inv;
  int i1 = 0;
#pragma unroll
  for (int e = 1; e < E_EXP; ++e) if (p[e] > p[i1]) i1 = e;
  int i2 = (i1 == 0) ? 1 : 0;
#pragma unroll
  for (int e = 0; e < E_EXP; ++e) if (e != i1 && p[e] > p[i2]) i2 = e;
  if (lane == 0) {
#pragma unroll
    for (int e = 0; e < E_EXP; ++e)
      combine[(size_t)t * E_EXP + e] = (e == i1 || e == i2) ? p[e] : 0.f;
  }
}

// ---------------- Up projection: H = gelu(x@w1^T) * (x@w3^T) (bf16 out) ----
__global__ __launch_bounds__(256) void moe_up_kernel(
    const float* __restrict__ x,
    const float* __restrict__ w1e,   // [I, D] row-major (this expert)
    const float* __restrict__ w3e,   // [I, D] row-major
    __bf16* __restrict__ H) {        // [T, I]
  __shared__ __align__(16) __bf16 Xs[TM * LDSK];
  __shared__ __align__(16) __bf16 W1s[TN * LDSK];
  __shared__ __align__(16) __bf16 W3s[TN * LDSK];

  const int tid  = threadIdx.x;
  const int lane = tid & 31;
  const int wave = tid >> 5;               // 0..7 -> M sub-tile
  const int m0 = blockIdx.y * TM;
  const int n0 = blockIdx.x * TN;

  v8f acc_a[TN / 16] = {};
  v8f acc_b[TN / 16] = {};

  for (int k0 = 0; k0 < D_DIM; k0 += TK) {
    // stage x tile [TM x TK] fp32 -> bf16 (1024 float4 chunks)
#pragma unroll
    for (int q = 0; q < 4; ++q) {
      const int idx4 = tid + 256 * q;
      const int row  = idx4 >> 3;
      const int col4 = (idx4 & 7) << 2;
      const float4 v = *(const float4*)&x[(size_t)(m0 + row) * D_DIM + k0 + col4];
      __bf16* dst = &Xs[row * LDSK + col4];
      dst[0] = (__bf16)v.x; dst[1] = (__bf16)v.y;
      dst[2] = (__bf16)v.z; dst[3] = (__bf16)v.w;
    }
    // stage w1/w3 tiles [TN x TK] fp32 -> bf16 (512 float4 chunks each)
#pragma unroll
    for (int q = 0; q < 2; ++q) {
      const int idx4 = tid + 256 * q;
      const int row  = idx4 >> 3;
      const int col4 = (idx4 & 7) << 2;
      const size_t g = (size_t)(n0 + row) * D_DIM + k0 + col4;
      const float4 a = *(const float4*)&w1e[g];
      const float4 b = *(const float4*)&w3e[g];
      __bf16* d1 = &W1s[row * LDSK + col4];
      d1[0] = (__bf16)a.x; d1[1] = (__bf16)a.y; d1[2] = (__bf16)a.z; d1[3] = (__bf16)a.w;
      __bf16* d3 = &W3s[row * LDSK + col4];
      d3[0] = (__bf16)b.x; d3[1] = (__bf16)b.y; d3[2] = (__bf16)b.z; d3[3] = (__bf16)b.w;
    }
    // prefetch next weight chunk into cache while we compute
    if (k0 + TK < D_DIM) {
      const size_t pg = (size_t)(n0 + (tid & 63)) * D_DIM + k0 + TK;
      __builtin_prefetch(&w1e[pg], 0, 1);
      __builtin_prefetch(&w3e[pg], 0, 1);
    }
    __syncthreads();

    const v16bf afrag = load_frag_b16(&Xs[(wave * 16) * LDSK], lane);
#pragma unroll
    for (int n = 0; n < TN / 16; ++n) {
      const v16bf b1 = load_frag_b16(&W1s[(n * 16) * LDSK], lane);
      const v16bf b3 = load_frag_b16(&W3s[(n * 16) * LDSK], lane);
      acc_a[n] = __builtin_amdgcn_wmma_f32_16x16x32_bf16(
          false, afrag, false, b1, (short)0, acc_a[n], false, false);
      acc_b[n] = __builtin_amdgcn_wmma_f32_16x16x32_bf16(
          false, afrag, false, b3, (short)0, acc_b[n], false, false);
    }
    __syncthreads();
  }

  // epilogue: exact gelu(a) * b -> bf16
  const int l  = lane & 15;
  const int hi = (lane >> 4) & 1;
#pragma unroll
  for (int n = 0; n < TN / 16; ++n) {
#pragma unroll
    for (int r = 0; r < 8; ++r) {
      const float a = acc_a[n][r];
      const float b = acc_b[n][r];
      const float g = 0.5f * a * (1.0f + erff(a * 0.70710678118654752f));
      const int m = m0 + wave * 16 + hi * 8 + r;
      H[(size_t)m * I_DIM + n0 + n * 16 + l] = (__bf16)(g * b);
    }
  }
}

// ---------------- Down projection + combine: out (+)= c[t,e] * (H @ w2^T) --
__global__ __launch_bounds__(256) void moe_down_kernel(
    const __bf16* __restrict__ H,     // [T, I] bf16
    const float* __restrict__ w2e,    // [D, I] row-major (this expert)
    const float* __restrict__ combine,
    float* __restrict__ out,          // [T, D] fp32
    int expert) {
  __shared__ __align__(16) __bf16 Hs[TM * LDSK];
  __shared__ __align__(16) __bf16 Ws[TN * LDSK];

  const int tid  = threadIdx.x;
  const int lane = tid & 31;
  const int wave = tid >> 5;
  const int m0 = blockIdx.y * TM;
  const int n0 = blockIdx.x * TN;

  v8f acc[TN / 16] = {};

  for (int k0 = 0; k0 < I_DIM; k0 += TK) {
    // stage H tile [TM x TK] bf16 via async global->LDS DMA (no VGPR round
    // trip, no conversion needed; tracked by ASYNCcnt, frees VALU/DS pipes
    // for the fp32->bf16 weight conversion below)
#pragma unroll
    for (int q = 0; q < 4; ++q) {
      const int idx  = tid + 256 * q;
      const int row  = idx >> 3;
      const int col4 = (idx & 7) << 2;
      const unsigned int lds_dst =
          (unsigned int)(size_t)&Hs[row * LDSK + col4];   // low 32b = LDS offset
      const __bf16* gsrc = &H[(size_t)(m0 + row) * I_DIM + k0 + col4];
      asm volatile("global_load_async_to_lds_b64 %0, %1, off"
                   :
                   : "v"(lds_dst), "v"(gsrc)
                   : "memory");
    }
    // stage w2 tile [TN x TK] fp32 -> bf16 (overlaps with async H staging)
#pragma unroll
    for (int q = 0; q < 2; ++q) {
      const int idx4 = tid + 256 * q;
      const int row  = idx4 >> 3;
      const int col4 = (idx4 & 7) << 2;
      const float4 a = *(const float4*)&w2e[(size_t)(n0 + row) * I_DIM + k0 + col4];
      __bf16* d = &Ws[row * LDSK + col4];
      d[0] = (__bf16)a.x; d[1] = (__bf16)a.y; d[2] = (__bf16)a.z; d[3] = (__bf16)a.w;
    }
    if (k0 + TK < I_DIM) {
      __builtin_prefetch(&w2e[(size_t)(n0 + (tid & 63)) * I_DIM + k0 + TK], 0, 1);
    }
    // async loads are tracked by ASYNCcnt (not DScnt): drain before barrier
    asm volatile("s_wait_asynccnt 0x0" ::: "memory");
    __syncthreads();

    const v16bf afrag = load_frag_b16(&Hs[(wave * 16) * LDSK], lane);
#pragma unroll
    for (int n = 0; n < TN / 16; ++n) {
      const v16bf bfrag = load_frag_b16(&Ws[(n * 16) * LDSK], lane);
      acc[n] = __builtin_amdgcn_wmma_f32_16x16x32_bf16(
          false, afrag, false, bfrag, (short)0, acc[n], false, false);
    }
    __syncthreads();
  }

  const int l  = lane & 15;
  const int hi = (lane >> 4) & 1;
  float scale[8];
#pragma unroll
  for (int r = 0; r < 8; ++r) {
    const int m = m0 + wave * 16 + hi * 8 + r;
    scale[r] = combine[(size_t)m * E_EXP + expert];
  }
#pragma unroll
  for (int n = 0; n < TN / 16; ++n) {
#pragma unroll
    for (int r = 0; r < 8; ++r) {
      const int m = m0 + wave * 16 + hi * 8 + r;
      const size_t o = (size_t)m * D_DIM + n0 + n * 16 + l;
      const float v = scale[r] * acc[n][r];
      out[o] = (expert == 0) ? v : (out[o] + v);
    }
  }
}

// ---------------------------------------------------------------------------
extern "C" void kernel_launch(void* const* d_in, const int* in_sizes, int n_in,
                              void* d_out, int out_size, void* d_ws, size_t ws_size,
                              hipStream_t stream) {
  (void)in_sizes; (void)n_in; (void)out_size; (void)ws_size;
  const float* x  = (const float*)d_in[0];   // [T, D]
  const float* wg = (const float*)d_in[1];   // [E, D]
  const float* w1 = (const float*)d_in[2];   // [E, I, D]
  const float* w3 = (const float*)d_in[3];   // [E, I, D]
  const float* w2 = (const float*)d_in[4];   // [E, D, I]
  float* out = (float*)d_out;                // [T, D]

  // workspace: combine [T,E] fp32, then H [T,I] bf16 (reused per expert)
  float* combine = (float*)d_ws;
  const size_t comb_bytes = ((size_t)T_TOK * E_EXP * sizeof(float) + 255) & ~(size_t)255;
  __bf16* H = (__bf16*)((char*)d_ws + comb_bytes);

  router_kernel<<<T_TOK / 8, 256, 0, stream>>>(x, wg, combine);

  const dim3 grid_up(I_DIM / TN, T_TOK / TM);
  const dim3 grid_dn(D_DIM / TN, T_TOK / TM);
  for (int e = 0; e < E_EXP; ++e) {
    const size_t wofs = (size_t)e * I_DIM * D_DIM;
    moe_up_kernel<<<grid_up, 256, 0, stream>>>(x, w1 + wofs, w3 + wofs, H);
    moe_down_kernel<<<grid_dn, 256, 0, stream>>>(H, w2 + (size_t)e * D_DIM * I_DIM,
                                                 combine, out, e);
  }
}
